// CrossAttentionBlock_57939108823366
// MI455X (gfx1250) — matrix operand order repaired and checked by reading, not compiled, not run on hardware
//
#include <hip/hip_runtime.h>

// ---------------------------------------------------------------------------
// CDNA5 (gfx1250) fused cross-attention block.
//   - all matmuls on V_WMMA_F32_16X16X32_F16 (wave32, f32 accumulate)
//   - K/V tile staging via Tensor Data Mover (tensor_load_to_lds) when the
//     toolchain exposes the builtin; manual staging fallback otherwise
// ---------------------------------------------------------------------------

typedef __attribute__((ext_vector_type(16))) _Float16     v16h;
typedef __attribute__((ext_vector_type(8)))  float        v8f;
typedef __attribute__((ext_vector_type(4)))  unsigned int u32x4;
typedef __attribute__((ext_vector_type(8)))  int          i32x8;
typedef __attribute__((ext_vector_type(4)))  int          i32x4;

#define Tt 3136   // packed tokens (16 * 196)
#define Dm 512    // model dim
#define Hh 8      // heads
#define HDm 64    // head dim

#if defined(__HIP_DEVICE_COMPILE__) && __has_builtin(__builtin_amdgcn_tensor_load_to_lds)
#define HAS_TDM 1
#endif

// Static segment boundaries from CHANNELS = [2,4,6,4] * S(=196)
__device__ __forceinline__ int seg_of(int t) {
    return (t < 392) ? 0 : (t < 1176) ? 1 : (t < 2352) ? 2 : 3;
}

__device__ __forceinline__ v8f wmma16(v16h a, v16h b, v8f c) {
    // (neg_a, A, neg_b, B, c_mod, C, reuse_a, reuse_b)
    return __builtin_amdgcn_wmma_f32_16x16x32_f16(false, a, false, b,
                                                  (short)0, c, false, false);
}

#if defined(HAS_TDM)
// TDM: DMA a 64x64 f16 tile (row stride 512 elems in memory) into LDS with
// 8-f16 row padding (LDS row stride 72 f16 = 144 B).
// D# per CDNA5 ISA:
//   group0: [1:0]=count=1, [63:32]=lds_addr, [120:64]=global_addr, [127:126]=2
//   group1: data_size=1(2B), pad_enable, pad_interval=4(32 dwords=one row),
//           pad_amount=3(4 dwords=16B), tensor_dim0=512, tensor_dim1=3136,
//           tile_dim0=64, tile_dim1=64, tensor_dim0_stride=512
__device__ __forceinline__ void tdm_load_tile(unsigned lds_off,
                                              const _Float16* gptr) {
    unsigned long long ga = (unsigned long long)(size_t)gptr;
    u32x4 g0 = { 1u, lds_off, (unsigned)ga,
                 (unsigned)(ga >> 32) | 0x80000000u };
    i32x8 g1 = { (int)((1u << 16) | (1u << 20) | (4u << 22) | (3u << 25)),
                 (int)(512u << 16),    // tensor_dim0[15:0] @ bits 63:48
                 (int)(3136u << 16),   // tensor_dim1[15:0] @ bits 95:80
                 (int)(64u << 16),     // tile_dim0 @ bits 127:112
                 64,                   // tile_dim1 @ bits 143:128
                 512,                  // tensor_dim0_stride @ bits 191:160
                 0, 0 };
    i32x4 gz = { 0, 0, 0, 0 };
#if __clang_major__ >= 23
    i32x8 gz8 = { 0, 0, 0, 0, 0, 0, 0, 0 };
    __builtin_amdgcn_tensor_load_to_lds(g0, g1, gz, gz, gz8, 0);
#else
    __builtin_amdgcn_tensor_load_to_lds(g0, g1, gz, gz, 0);
#endif
}
#endif

// ---------------------------------------------------------------------------
// GEMM:  Y[T,512] = cast<OutT>( (X (+P)) @ W + bias )
// Tile: BM=64, BN=64, BK=32; 256 threads = 8 waves in a 4(m) x 2(n) grid.
// Weight tile stored transposed in LDS so B-fragments read contiguous b128.
// ---------------------------------------------------------------------------
template <bool ADD_POS, typename InT, typename OutT>
__global__ __launch_bounds__(256)
void proj_gemm(const InT* __restrict__ X, const float* __restrict__ P,
               const float* __restrict__ W, const float* __restrict__ bias,
               OutT* __restrict__ Y)
{
    __shared__ __align__(16) _Float16 As[64][40];    // [row][k]  (+pad)
    __shared__ __align__(16) _Float16 BsT[64][40];   // [col][k]  (+pad)

    const int tid  = threadIdx.x;
    const int bm   = blockIdx.x, bn = blockIdx.y;
    const int lane = tid & 31,   wid = tid >> 5;
    const int half = lane >> 4;          // lane group 0..1
    const int m    = lane & 15;          // A-row / B-col within fragment
    const int wm   = wid & 3, wn = wid >> 2;

    v8f z = {};
    v8f acc[2]; acc[0] = z; acc[1] = z;

    // LDS fill mapping: 8 contiguous source elements per thread
    const int ar = (tid * 8) / 32, ac = (tid * 8) % 32;   // A: 64x32
    const int br = (tid * 8) / 64, bc = (tid * 8) % 64;   // B: 32x64

    for (int kk = 0; kk < 16; ++kk) {          // K = 512 / BK(32)
        __syncthreads();
        {
            const InT*   src  = X + (size_t)(bm * 64 + ar) * Dm + kk * 32 + ac;
            const float* wsrc = W + (size_t)(kk * 32 + br) * Dm + bn * 64 + bc;
            if (kk < 15) {                      // gfx1250 global_prefetch_b8
                __builtin_prefetch(src + 32, 0, 0);
                __builtin_prefetch(wsrc + 32 * Dm, 0, 0);
            }
            #pragma unroll
            for (int j = 0; j < 8; ++j) {
                float xv = (float)src[j];
                if constexpr (ADD_POS)
                    xv += P[(size_t)(bm * 64 + ar) * Dm + kk * 32 + ac + j];
                As[ar][ac + j] = (_Float16)xv;
            }
            #pragma unroll
            for (int j = 0; j < 8; ++j)          // transpose on fill
                BsT[bc + j][br] = (_Float16)wsrc[j];
        }
        __syncthreads();

        // A-fragment (16x32 f16): M=m, K = half*8+i and 16+half*8+i
        v16h a;
        #pragma unroll
        for (int i = 0; i < 8; ++i) {
            a[i]     = As[wm * 16 + m][half * 8 + i];
            a[8 + i] = As[wm * 16 + m][16 + half * 8 + i];
        }
        #pragma unroll
        for (int nt = 0; nt < 2; ++nt) {
            // B-fragment (32x16): N=m, K = half*8+j and 16+half*8+(j-8)
            v16h b;
            #pragma unroll
            for (int j = 0; j < 16; ++j)
                b[j] = BsT[wn * 32 + nt * 16 + m][half * 8 + (j & 7) + ((j >> 3) << 4)];
            acc[nt] = wmma16(a, b, acc[nt]);
        }
    }

    // C layout: row = half*8 + i, col = lane&15
    #pragma unroll
    for (int nt = 0; nt < 2; ++nt) {
        #pragma unroll
        for (int i = 0; i < 8; ++i) {
            const int row = bm * 64 + wm * 16 + half * 8 + i;
            const int col = bn * 64 + wn * 32 + nt * 16 + m;
            float v = acc[nt][i] + bias[col];
            Y[(size_t)row * Dm + col] = (OutT)v;
        }
    }
}

// ---------------------------------------------------------------------------
// Flash attention over packed segments.
// Block = (q-tile of 64 rows, head); 128 threads = 4 waves, 16 q-rows/wave.
// Per 64-key tile: TDM-stage K/V -> QK^T (8 wmma) -> mask -> online softmax
// -> P@V (8 wmma).
// ---------------------------------------------------------------------------
__global__ __launch_bounds__(128)
void attn_kernel(const _Float16* __restrict__ Q, const _Float16* __restrict__ K,
                 const _Float16* __restrict__ V, _Float16* __restrict__ O)
{
    __shared__ __align__(16) _Float16 Kt[64][72];    // [key][dim]
    __shared__ __align__(16) _Float16 Vt[64][72];    // [key][dim]
    __shared__ __align__(16) _Float16 Pl[4][16][64]; // per-wave P relayout

    const int tid  = threadIdx.x;
    const int lane = tid & 31, wave = tid >> 5;
    const int half = lane >> 4, m = lane & 15;
    const int qt = blockIdx.x, h = blockIdx.y;

    // Preload Q A-fragments (M = q row, K = head dim, 2 steps of 32)
    const int qrowA = qt * 64 + wave * 16 + m;
    v16h qf[2];
    #pragma unroll
    for (int ks = 0; ks < 2; ++ks) {
        #pragma unroll
        for (int i = 0; i < 8; ++i) {
            qf[ks][i]     = Q[(size_t)qrowA * Dm + h * HDm + ks * 32 + half * 8 + i];
            qf[ks][8 + i] = Q[(size_t)qrowA * Dm + h * HDm + ks * 32 + 16 + half * 8 + i];
        }
    }

    // q-row segment ids are kt-invariant: hoist out of the key loop
    int qseg[8];
    #pragma unroll
    for (int i = 0; i < 8; ++i)
        qseg[i] = seg_of(qt * 64 + wave * 16 + half * 8 + i);

    float mrow[8], lrow[8];
    v8f z = {};
    v8f o[4];
    #pragma unroll
    for (int i = 0; i < 8; ++i) { mrow[i] = -3.0e38f; lrow[i] = 0.0f; }
    #pragma unroll
    for (int nt = 0; nt < 4; ++nt) o[nt] = z;

    const int ldr = tid >> 1;             // fallback staging: row 0..63
    const int ldc = (tid & 1) * 32;       // col 0 or 32
    const float scale = 0.125f;           // 1/sqrt(64)

    for (int kt = 0; kt < Tt / 64; ++kt) {
        __syncthreads();                  // all reads of previous tile done
#if defined(HAS_TDM)
        if (wave == 0) {
            // TDM issues once per wave (EXEC ignored): wave 0 only.
            tdm_load_tile((unsigned)(size_t)&Kt[0][0],
                          K + (size_t)(kt * 64) * Dm + h * HDm);
            tdm_load_tile((unsigned)(size_t)&Vt[0][0],
                          V + (size_t)(kt * 64) * Dm + h * HDm);
            __builtin_amdgcn_s_wait_tensorcnt(0);
        }
#else
        #pragma unroll
        for (int j = 0; j < 32; ++j) {
            Kt[ldr][ldc + j] = K[(size_t)(kt * 64 + ldr) * Dm + h * HDm + ldc + j];
            Vt[ldr][ldc + j] = V[(size_t)(kt * 64 + ldr) * Dm + h * HDm + ldc + j];
        }
#endif
        __syncthreads();                  // tile visible to all waves

        // ---- S = Q K^T (contraction over head dim; B rows contiguous) ----
        v8f s[4];
        #pragma unroll
        for (int nt = 0; nt < 4; ++nt) {
            s[nt] = z;
            #pragma unroll
            for (int ks = 0; ks < 2; ++ks) {
                v16h b;   // B[d, key] = Kt[key][d]
                #pragma unroll
                for (int j = 0; j < 16; ++j)
                    b[j] = Kt[nt * 16 + m][ks * 32 + half * 8 + (j & 7) + ((j >> 3) << 4)];
                s[nt] = wmma16(qf[ks], b, s[nt]);
            }
        }

        // ---- scale + block-diagonal segment mask ----
        #pragma unroll
        for (int nt = 0; nt < 4; ++nt) {
            const int ksec = seg_of(kt * 64 + nt * 16 + m);
            #pragma unroll
            for (int i = 0; i < 8; ++i) {
                const float sv = s[nt][i] * scale;
                s[nt][i] = (qseg[i] == ksec) ? sv : -1.0e30f;
            }
        }

        // ---- online softmax (rows live across 16-lane half-waves) ----
        #pragma unroll
        for (int i = 0; i < 8; ++i) {
            float mx = -3.0e38f;
            #pragma unroll
            for (int nt = 0; nt < 4; ++nt) mx = fmaxf(mx, s[nt][i]);
            #pragma unroll
            for (int off = 1; off < 16; off <<= 1)
                mx = fmaxf(mx, __shfl_xor(mx, off, 32));
            const float mnew  = fmaxf(mrow[i], mx);
            const float alpha = __expf(mrow[i] - mnew);
            float rs = 0.0f;
            #pragma unroll
            for (int nt = 0; nt < 4; ++nt) {
                const float p = __expf(s[nt][i] - mnew);
                s[nt][i] = p;
                rs += p;
            }
            #pragma unroll
            for (int off = 1; off < 16; off <<= 1)
                rs += __shfl_xor(rs, off, 32);
            lrow[i] = lrow[i] * alpha + rs;
            mrow[i] = mnew;
            #pragma unroll
            for (int nt = 0; nt < 4; ++nt) o[nt][i] *= alpha;
        }

        // ---- P: C-fragment layout -> LDS -> A-fragment layout ----
        #pragma unroll
        for (int nt = 0; nt < 4; ++nt)
            #pragma unroll
            for (int i = 0; i < 8; ++i)
                Pl[wave][half * 8 + i][nt * 16 + m] = (_Float16)s[nt][i];
        __syncthreads();

        // ---- O += P @ V (contraction over 64 keys, 2 steps of 32) ----
        #pragma unroll
        for (int ks = 0; ks < 2; ++ks) {
            v16h a;
            #pragma unroll
            for (int i = 0; i < 8; ++i) {
                a[i]     = Pl[wave][m][ks * 32 + half * 8 + i];
                a[8 + i] = Pl[wave][m][ks * 32 + 16 + half * 8 + i];
            }
            #pragma unroll
            for (int nt = 0; nt < 4; ++nt) {
                v16h b;   // B[key, dim] = Vt[key][dim]
                #pragma unroll
                for (int j = 0; j < 16; ++j)
                    b[j] = Vt[ks * 32 + half * 8 + (j & 7) + ((j >> 3) << 4)][nt * 16 + m];
                o[nt] = wmma16(a, b, o[nt]);
            }
        }
    }

    // ---- normalize and write head output (f16, packed [T, H*HD]) ----
    #pragma unroll
    for (int nt = 0; nt < 4; ++nt) {
        #pragma unroll
        for (int i = 0; i < 8; ++i) {
            const int row = qt * 64 + wave * 16 + half * 8 + i;
            const int col = h * HDm + nt * 16 + m;
            O[(size_t)row * Dm + col] = (_Float16)(o[nt][i] / lrow[i]);
        }
    }
}

// ---------------------------------------------------------------------------
extern "C" void kernel_launch(void* const* d_in, const int* in_sizes, int n_in,
                              void* d_out, int out_size, void* d_ws, size_t ws_size,
                              hipStream_t stream)
{
    (void)in_sizes; (void)n_in; (void)out_size; (void)ws_size;

    const float* xq  = (const float*)d_in[0];   // [16,196,512]
    const float* xk  = (const float*)d_in[1];
    const float* pos = (const float*)d_in[2];
    // d_in[3] = channels [2,4,6,4] -> static, baked into seg_of()
    const float* Wq = (const float*)d_in[4];
    const float* bq = (const float*)d_in[5];
    const float* Wk = (const float*)d_in[6];
    const float* bk = (const float*)d_in[7];
    const float* Wv = (const float*)d_in[8];
    const float* bv = (const float*)d_in[9];
    const float* Wo = (const float*)d_in[10];
    const float* bo = (const float*)d_in[11];

    // Workspace: Q, K, V, attn_out as f16 [T,512]  (~12.8 MB total)
    _Float16* Qw = (_Float16*)d_ws;
    _Float16* Kw = Qw + (size_t)Tt * Dm;
    _Float16* Vw = Kw + (size_t)Tt * Dm;
    _Float16* Aw = Vw + (size_t)Tt * Dm;

    dim3 g(Tt / 64, Dm / 64);   // 49 x 8 tiles

    proj_gemm<true,  float,    _Float16><<<g, 256, 0, stream>>>(xq, pos, Wq, bq, Qw);
    proj_gemm<true,  float,    _Float16><<<g, 256, 0, stream>>>(xk, pos, Wk, bk, Kw);
    proj_gemm<false, float,    _Float16><<<g, 256, 0, stream>>>(xk, nullptr, Wv, bv, Vw);

    attn_kernel<<<dim3(Tt / 64, Hh), 128, 0, stream>>>(Qw, Kw, Vw, Aw);

    proj_gemm<false, _Float16, float   ><<<g, 256, 0, stream>>>(Aw, nullptr, Wo, bo,
                                                                (float*)d_out);
}
